// TransformerBlock_28870770163781
// MI455X (gfx1250) — matrix operand order repaired and checked by reading, not compiled, not run on hardware
//
#include <hip/hip_runtime.h>
#include <hip/hip_bf16.h>
#include <math.h>

// Problem constants (B,S,D,H from reference)
#define Bn  4
#define Sn  2048
#define Dn  512
#define Hn  8
#define HDn 64
#define Tn  (Bn * Sn)   // 8192 tokens

typedef _Float16 v16h __attribute__((ext_vector_type(16)));
typedef _Float16 v8h  __attribute__((ext_vector_type(8)));
typedef float    v8f  __attribute__((ext_vector_type(8)));
typedef unsigned int u32x4 __attribute__((ext_vector_type(4)));
typedef unsigned int u32x8 __attribute__((ext_vector_type(8)));

enum { OUT_F32 = 1, OUT_F16 = 2, OUT_VT = 4, DO_RELU = 8, ADD_RES = 16 };

__device__ __forceinline__ v8f wmma16(v16h a, v16h b, v8f c) {
  // D = A(16x32 f16) * B(32x16 f16) + C(16x16 f32)
  return __builtin_amdgcn_wmma_f32_16x16x32_f16(false, a, false, b, (short)0, c,
                                                false, false);
}

// async global -> LDS copy, 16B per lane, tracked by ASYNCcnt
__device__ __forceinline__ void async_copy_b128(const _Float16* gsrc,
                                                const _Float16* lds_dst) {
  unsigned la = (unsigned)(unsigned long long)lds_dst;  // low 32 bits = LDS offset
  asm volatile("global_load_async_to_lds_b128 %0, %1, off"
               :: "v"(la), "v"(gsrc) : "memory");
}

__device__ __forceinline__ void async_wait0() {
  asm volatile("s_wait_asynccnt 0" ::: "memory");
}

// Tensor Data Mover: DMA a 2D tile (tile1 rows x tile0 elems of 2B, row stride
// stride0 elems) from global memory into LDS, packed row-major contiguous.
// D# group0/group1 per CDNA5 ISA 8.3/8.4; VADDR2/3 = NULL (<=2D tensor).
// Tracked by per-wave TENSORcnt.
__device__ __forceinline__ void tdm_load_2d(unsigned lds_addr, const void* gaddr,
                                            unsigned tile0, unsigned tile1,
                                            unsigned tdim0, unsigned tdim1,
                                            unsigned stride0) {
  unsigned long long ga = (unsigned long long)gaddr;
  u32x4 g0;
  g0[0] = 1u;  // count=1 user descriptor; no gather/restore
  g0[1] = (unsigned)__builtin_amdgcn_readfirstlane((int)lds_addr);
  g0[2] = (unsigned)__builtin_amdgcn_readfirstlane((int)(unsigned)ga);
  g0[3] = ((unsigned)__builtin_amdgcn_readfirstlane((int)(unsigned)(ga >> 32)) &
           0x01FFFFFFu) | 0x80000000u;  // global_addr[56:32] | type=2
  u32x8 g1;
  g1[0] = 0x00010000u;                    // wg_mask=0, data_size=1 (2 bytes)
  g1[1] = tdim0 << 16;                    // tensor_dim0[15:0]
  g1[2] = (tdim0 >> 16) | (tdim1 << 16);  // tensor_dim0[31:16] | tensor_dim1[15:0]
  g1[3] = (tdim1 >> 16) | (tile0 << 16);  // tensor_dim1[31:16] | tile_dim0
  g1[4] = tile1;                          // tile_dim1 (tile_dim2 = 0)
  g1[5] = stride0;                        // tensor_dim0_stride[31:0]
  g1[6] = 0u;                             // stride0[47:32]=0, dim1_stride=0
  g1[7] = 0u;
  asm volatile("tensor_load_to_lds %0, %1" :: "s"(g0), "s"(g1) : "memory");
}

// ---------------- conversion kernels ----------------
__global__ void cvt_f16(const float* __restrict__ in, _Float16* __restrict__ out, int n) {
  int i = blockIdx.x * blockDim.x + threadIdx.x;
  if (i < n) out[i] = (_Float16)in[i];
}

// wt[n][k] = w[k][n]  (transpose so WMMA B-frags are contiguous per lane)
__global__ void cvt_f16_T(const float* __restrict__ w, _Float16* __restrict__ wt, int d) {
  int i = blockIdx.x * blockDim.x + threadIdx.x;
  if (i < d * d) {
    int nn = i / d, kk = i - nn * d;
    wt[i] = (_Float16)w[(size_t)kk * d + nn];
  }
}

// ---------------- generic WMMA GEMM with async-LDS weight staging ----------------
// One wave computes a 16(M) x 64(N) strip; the block shares one 64-col weight
// strip, staged 32-K at a time into LDS via global_load_async_to_lds_b128
// (double-buffered: DMA for step i+1 overlaps WMMAs of step i).
// A:[M][K] f16 row-major, Bt:[N][K] f16 row-major (B transposed).
__global__ __launch_bounds__(256) void gemm16(
    const _Float16* __restrict__ A, const _Float16* __restrict__ Bt,
    const float* __restrict__ bias, const float* __restrict__ resid,
    float* __restrict__ outF, _Float16* __restrict__ outH,
    _Float16* __restrict__ outVT, int M, int N, int K, int flags) {
  const int lane = threadIdx.x & 31;
  const int r = lane & 15;          // col (B/C) or row (A) selector
  const int g = lane >> 4;          // half-wave group
  const int klo = g * 8;            // unused; A-frag K base is g * 16
  const int tm = blockIdx.x * 8 + threadIdx.y;   // 16-row M tile
  const int nb = blockIdx.y * 64;                // 64-col N strip (shared in block)
  const int row0 = tm * 16;

  __shared__ __align__(64) _Float16 bstage[2][64][32];

  // staging map: wave wv covers rows [wv*8, wv*8+8), 16B chunk per lane
  const int rr = threadIdx.y * 8 + (lane >> 2);  // 0..63
  const int c4 = (lane & 3) * 8;                 // half-offset 0/8/16/24
  const _Float16* bsrc = Bt + (size_t)(nb + rr) * K + c4;

  // prologue: stage k0 = 0 into buffer 0
  async_copy_b128(bsrc, &bstage[0][rr][c4]);

  v8f acc[4];
#pragma unroll
  for (int j = 0; j < 4; ++j) {
    float bval = bias ? bias[nb + j * 16 + r] : 0.0f;
#pragma unroll
    for (int v = 0; v < 8; ++v) acc[j][v] = bval;
  }

  const _Float16* arow = A + (size_t)(row0 + r) * K;
  for (int k0 = 0; k0 < K; k0 += 32) {
    const int buf = (k0 >> 5) & 1;
    async_wait0();       // own slice of bstage[buf] landed
    __syncthreads();     // all slices visible; prior reads of buf^1 retired
    if (k0 + 32 < K)     // overlap next DMA with this step's WMMAs
      async_copy_b128(bsrc + k0 + 32, &bstage[buf ^ 1][rr][c4]);

    // A fragment: lane holds row (l&15), K = g*16 + e (contiguous)
    union { v16h v; v8h h[2]; } ua;
    ua.h[0] = *(const v8h*)(arow + k0 + g * 16);
    ua.h[1] = *(const v8h*)(arow + k0 + g * 16 + 8);
    if (k0 + 32 < K) __builtin_prefetch(arow + k0 + 32, 0, 3);

    // B fragments from LDS: lane holds col (l&15), K = g*16 + e (contiguous)
    v16h bb[4];
#pragma unroll
    for (int j = 0; j < 4; ++j)
      bb[j] = *(const v16h*)&bstage[buf][j * 16 + r][g * 16];
#pragma unroll
    for (int j = 0; j < 4; ++j)
      acc[j] = wmma16(ua.v, bb[j], acc[j]);
  }

  // Epilogue: C layout => element v maps to row = row0 + v + 8g, col = nb + j*16 + r
#pragma unroll
  for (int j = 0; j < 4; ++j) {
#pragma unroll
    for (int v = 0; v < 8; ++v) {
      int row = row0 + v + g * 8;
      int col = nb + j * 16 + r;
      float val = acc[j][v];
      if (flags & DO_RELU) val = fmaxf(val, 0.0f);
      if (flags & ADD_RES) val += resid[(size_t)row * N + col];
      if (flags & OUT_F32) outF[(size_t)row * N + col] = val;
      if (flags & OUT_F16) outH[(size_t)row * N + col] = (_Float16)val;
      if (flags & OUT_VT) {
        // store V transposed per head: Vt[b][h][d][s]
        int bb2 = row >> 11, ss = row & (Sn - 1);
        int hh = col >> 6, dd = col & (HDn - 1);
        outVT[(((size_t)bb2 * Hn + hh) * HDn + dd) * Sn + ss] = (_Float16)val;
      }
    }
  }
}

// ---------------- flash attention with TDM K-tile staging ----------------
// One wave: one (b, h, 16-query) tile. The wave's 32-key K tile (32 rows x
// 64 halves, row stride D) is DMA'd into LDS by the Tensor Data Mover,
// double-buffered on the per-wave TENSORcnt. Online softmax over 32-key
// blocks bounded by the causal limit; P goes C-layout -> A-layout via LDS.
__global__ __launch_bounds__(256) void attn_fa(
    const _Float16* __restrict__ Qh, const _Float16* __restrict__ Kh,
    const _Float16* __restrict__ Vt, _Float16* __restrict__ ctxh) {
  const int lane = threadIdx.x & 31;
  const int wv = threadIdx.y;
  const int wid = blockIdx.x * 8 + wv;
  const int QT = Sn / 16;                 // 128 query tiles per (b,h)
  const int qi = wid % QT;
  const int bh = wid / QT;
  const int b = bh / Hn, h = bh % Hn;
  const int r = lane & 15, g = lane >> 4, klo = g * 8;
  const int t0 = qi * 16;

  __shared__ __align__(16) _Float16 plds[8][16][32];
  __shared__ __align__(64) _Float16 kst[8][2][32][64];   // per-wave K tiles

  const _Float16* kbase = Kh + (size_t)b * Sn * Dn + h * HDn;
  unsigned kst_lds[2];
  kst_lds[0] = (unsigned)(unsigned long long)&kst[wv][0][0][0];
  kst_lds[1] = (unsigned)(unsigned long long)&kst[wv][1][0][0];

  // prologue: TDM stage keys [0,32) into buffer 0
  tdm_load_2d(kst_lds[0], kbase, /*tile0=*/HDn, /*tile1=*/32,
              /*tdim0=*/HDn, /*tdim1=*/32, /*stride0=*/Dn);

  // Q A-fragments for the two 32-wide d-steps (HD = 64)
  v16h aq[2];
#pragma unroll
  for (int ks = 0; ks < 2; ++ks) {
    const _Float16* qp = Qh + ((size_t)b * Sn + t0 + r) * Dn + h * HDn + ks * 32 + g * 16;
    union { v16h v; v8h hh[2]; } u;
    u.hh[0] = *(const v8h*)qp;
    u.hh[1] = *(const v8h*)(qp + 8);
    aq[ks] = u.v;
  }

  float mrow[8], lrow[8];
#pragma unroll
  for (int v = 0; v < 8; ++v) { mrow[v] = -3.0e38f; lrow[v] = 0.0f; }
  v8f o[4];
#pragma unroll
  for (int j = 0; j < 4; ++j)
#pragma unroll
    for (int v = 0; v < 8; ++v) o[j][v] = 0.0f;

  const int kkend = t0 + 15;  // last causally-visible key index
  for (int kk2 = 0; kk2 <= kkend; kk2 += 32) {
    const int buf = (kk2 >> 5) & 1;
    // issue next tile's DMA (overlaps this step's compute), wait for current
    if (kk2 + 32 <= kkend) {
      tdm_load_2d(kst_lds[buf ^ 1], kbase + (size_t)(kk2 + 32) * Dn,
                  HDn, 32, HDn, 32, Dn);
      __builtin_amdgcn_s_wait_tensorcnt(1);   // in-order: oldest (buf) done
    } else {
      __builtin_amdgcn_s_wait_tensorcnt(0);
    }
    asm volatile("" ::: "memory");

    // --- scores: two 16x16 sub-tiles over keys [kk2, kk2+32) ---
    v8f sc[2];
#pragma unroll
    for (int c = 0; c < 2; ++c) {
      v8f s;
#pragma unroll
      for (int v = 0; v < 8; ++v) s[v] = 0.0f;
      int kb = kk2 + c * 16;
#pragma unroll
      for (int ks = 0; ks < 2; ++ks) {
        // B-frag from the TDM-staged LDS tile: 32B contiguous per lane
        v16h bb = *(const v16h*)&kst[wv][buf][c * 16 + r][ks * 32 + g * 16];
        s = wmma16(aq[ks], bb, s);
      }
#pragma unroll
      for (int v = 0; v < 8; ++v) {
        int row = t0 + v + g * 8;
        int col = kb + r;
        float val = s[v] * 0.125f;           // 1/sqrt(HD)
        if (col > row) val = -1.0e9f;        // causal mask (same as reference)
        s[v] = val;
      }
      sc[c] = s;
    }

    // --- online softmax (butterfly over the 16-lane row groups) ---
    float alpha[8];
#pragma unroll
    for (int v = 0; v < 8; ++v) {
      float lm = fmaxf(sc[0][v], sc[1][v]);
#pragma unroll
      for (int off = 1; off < 16; off <<= 1) lm = fmaxf(lm, __shfl_xor(lm, off, 32));
      float mnew = fmaxf(mrow[v], lm);
      alpha[v] = __expf(mrow[v] - mnew);
      float p0 = __expf(sc[0][v] - mnew);
      float p1 = __expf(sc[1][v] - mnew);
      sc[0][v] = p0; sc[1][v] = p1;
      float rs = p0 + p1;
#pragma unroll
      for (int off = 1; off < 16; off <<= 1) rs += __shfl_xor(rs, off, 32);
      lrow[v] = lrow[v] * alpha[v] + rs;
      mrow[v] = mnew;
    }
#pragma unroll
    for (int j = 0; j < 4; ++j)
#pragma unroll
      for (int v = 0; v < 8; ++v) o[j][v] *= alpha[v];

    // --- P: C-layout -> A-layout through per-wave LDS tile ---
#pragma unroll
    for (int v = 0; v < 8; ++v) {
      int rr = v + g * 8;
      plds[wv][rr][r]      = (_Float16)sc[0][v];
      plds[wv][rr][16 + r] = (_Float16)sc[1][v];
    }
    asm volatile("s_wait_dscnt 0" ::: "memory");
    __builtin_amdgcn_wave_barrier();
    union { v16h v; v8h hh[2]; } up;
    up.hh[0] = *(const v8h*)&plds[wv][r][g * 16];
    up.hh[1] = *(const v8h*)&plds[wv][r][g * 16 + 8];
    v16h pa = up.v;
    __builtin_amdgcn_wave_barrier();

    // --- ctx += P * V  (B-frag from pre-transposed V, contiguous per lane) ---
#pragma unroll
    for (int j = 0; j < 4; ++j) {
      const _Float16* vp = Vt + (((size_t)b * Hn + h) * HDn + j * 16 + r) * Sn + kk2 + g * 16;
      v16h bb = *(const v16h*)vp;
      o[j] = wmma16(pa, bb, o[j]);
    }
  }

  // normalize and store ctx (f16, [T][D] layout, per-head columns)
#pragma unroll
  for (int v = 0; v < 8; ++v) {
    float inv = 1.0f / lrow[v];
    int row = t0 + v + g * 8;
#pragma unroll
    for (int j = 0; j < 4; ++j) {
      ctxh[((size_t)b * Sn + row) * Dn + h * HDn + j * 16 + r] = (_Float16)(o[j][v] * inv);
    }
  }
}

// ---------------- layernorm over D=512 (one block per row) ----------------
__global__ __launch_bounds__(256) void ln512(
    const float* __restrict__ in, const float* __restrict__ gam,
    const float* __restrict__ bet, float* __restrict__ out,
    _Float16* __restrict__ outH) {
  const int row = blockIdx.x;
  const float* p = in + (size_t)row * Dn;
  __shared__ float red[8];
  __shared__ float stat[2];
  float v0 = p[threadIdx.x], v1 = p[threadIdx.x + 256];
  float s = v0 + v1;
#pragma unroll
  for (int off = 16; off; off >>= 1) s += __shfl_xor(s, off, 32);
  if ((threadIdx.x & 31) == 0) red[threadIdx.x >> 5] = s;
  __syncthreads();
  if (threadIdx.x == 0) {
    float t = 0;
    for (int i = 0; i < 8; ++i) t += red[i];
    stat[0] = t / (float)Dn;
  }
  __syncthreads();
  float mean = stat[0];
  float d0 = v0 - mean, d1 = v1 - mean;
  float vs = d0 * d0 + d1 * d1;
#pragma unroll
  for (int off = 16; off; off >>= 1) vs += __shfl_xor(vs, off, 32);
  if ((threadIdx.x & 31) == 0) red[threadIdx.x >> 5] = vs;
  __syncthreads();
  if (threadIdx.x == 0) {
    float t = 0;
    for (int i = 0; i < 8; ++i) t += red[i];
    stat[1] = rsqrtf(t / (float)Dn + 1e-5f);
  }
  __syncthreads();
  float rstd = stat[1];
  int c0 = threadIdx.x, c1 = threadIdx.x + 256;
  float o0 = d0 * rstd * gam[c0] + bet[c0];
  float o1 = d1 * rstd * gam[c1] + bet[c1];
  out[(size_t)row * Dn + c0] = o0;
  out[(size_t)row * Dn + c1] = o1;
  if (outH) {
    outH[(size_t)row * Dn + c0] = (_Float16)o0;
    outH[(size_t)row * Dn + c1] = (_Float16)o1;
  }
}

// ---------------- host orchestration ----------------
extern "C" void kernel_launch(void* const* d_in, const int* in_sizes, int n_in,
                              void* d_out, int out_size, void* d_ws, size_t ws_size,
                              hipStream_t stream) {
  (void)in_sizes; (void)n_in; (void)out_size; (void)ws_size;
  const float* x    = (const float*)d_in[0];
  // d_in[1] = mask (unused: causal mask computed analytically)
  const float* wq   = (const float*)d_in[2];
  const float* bq   = (const float*)d_in[3];
  const float* wk   = (const float*)d_in[4];
  const float* bk   = (const float*)d_in[5];
  const float* wv   = (const float*)d_in[6];
  const float* bv   = (const float*)d_in[7];
  const float* wo   = (const float*)d_in[8];
  const float* bo   = (const float*)d_in[9];
  const float* ln1g = (const float*)d_in[10];
  const float* ln1b = (const float*)d_in[11];
  const float* ln2g = (const float*)d_in[12];
  const float* ln2b = (const float*)d_in[13];
  const float* w1   = (const float*)d_in[14];
  const float* b1   = (const float*)d_in[15];
  const float* w2   = (const float*)d_in[16];
  const float* b2   = (const float*)d_in[17];
  float* out = (float*)d_out;

  char* ws = (char*)d_ws;
  size_t off = 0;
  auto alloc = [&](size_t bytes) {
    char* p = ws + off;
    off += (bytes + 255) & ~(size_t)255;
    return p;
  };
  _Float16* xh   = (_Float16*)alloc((size_t)Tn * Dn * 2);
  _Float16* wqt  = (_Float16*)alloc((size_t)Dn * Dn * 2);
  _Float16* wkt  = (_Float16*)alloc((size_t)Dn * Dn * 2);
  _Float16* wvt  = (_Float16*)alloc((size_t)Dn * Dn * 2);
  _Float16* wot  = (_Float16*)alloc((size_t)Dn * Dn * 2);
  _Float16* w1t  = (_Float16*)alloc((size_t)Dn * Dn * 2);
  _Float16* w2t  = (_Float16*)alloc((size_t)Dn * Dn * 2);
  _Float16* Qh   = (_Float16*)alloc((size_t)Tn * Dn * 2);
  _Float16* Kh   = (_Float16*)alloc((size_t)Tn * Dn * 2);
  _Float16* Vt   = (_Float16*)alloc((size_t)Tn * Dn * 2);
  _Float16* ctxh = (_Float16*)alloc((size_t)Tn * Dn * 2);
  float*    r1   = (float*)alloc((size_t)Tn * Dn * 4);
  float*    x1   = (float*)alloc((size_t)Tn * Dn * 4);
  _Float16* x1h  = (_Float16*)alloc((size_t)Tn * Dn * 2);
  _Float16* h1h  = Qh;   // Q dead after attention: reuse
  float*    r2   = r1;   // r1 dead after LN1: reuse

  dim3 blkG(32, 8);
  dim3 ggrid(Tn / 128, Dn / 64);

  // f16 conversions (+ weight transposes)
  cvt_f16<<<(Tn * Dn) / 256, 256, 0, stream>>>(x, xh, Tn * Dn);
  cvt_f16_T<<<(Dn * Dn) / 256, 256, 0, stream>>>(wq, wqt, Dn);
  cvt_f16_T<<<(Dn * Dn) / 256, 256, 0, stream>>>(wk, wkt, Dn);
  cvt_f16_T<<<(Dn * Dn) / 256, 256, 0, stream>>>(wv, wvt, Dn);
  cvt_f16_T<<<(Dn * Dn) / 256, 256, 0, stream>>>(wo, wot, Dn);
  cvt_f16_T<<<(Dn * Dn) / 256, 256, 0, stream>>>(w1, w1t, Dn);
  cvt_f16_T<<<(Dn * Dn) / 256, 256, 0, stream>>>(w2, w2t, Dn);

  // QKV projections
  gemm16<<<ggrid, blkG, 0, stream>>>(xh, wqt, bq, nullptr, nullptr, Qh, nullptr,
                                     Tn, Dn, Dn, OUT_F16);
  gemm16<<<ggrid, blkG, 0, stream>>>(xh, wkt, bk, nullptr, nullptr, Kh, nullptr,
                                     Tn, Dn, Dn, OUT_F16);
  gemm16<<<ggrid, blkG, 0, stream>>>(xh, wvt, bv, nullptr, nullptr, nullptr, Vt,
                                     Tn, Dn, Dn, OUT_VT);

  // flash attention (TDM-staged K tiles)
  attn_fa<<<(Bn * Hn * (Sn / 16)) / 8, blkG, 0, stream>>>(Qh, Kh, Vt, ctxh);

  // output projection + residual, then LN1
  gemm16<<<ggrid, blkG, 0, stream>>>(ctxh, wot, bo, x, r1, nullptr, nullptr,
                                     Tn, Dn, Dn, ADD_RES | OUT_F32);
  ln512<<<Tn, 256, 0, stream>>>(r1, ln1g, ln1b, x1, x1h);

  // FFN: relu(x1 @ w1 + b1), relu(h1 @ w2 + b2) + x1, LN2 -> out
  gemm16<<<ggrid, blkG, 0, stream>>>(x1h, w1t, b1, nullptr, nullptr, h1h, nullptr,
                                     Tn, Dn, Dn, DO_RELU | OUT_F16);
  gemm16<<<ggrid, blkG, 0, stream>>>(h1h, w2t, b2, x1, r2, nullptr, nullptr,
                                     Tn, Dn, Dn, DO_RELU | ADD_RES | OUT_F32);
  ln512<<<Tn, 256, 0, stream>>>(r2, ln2g, ln2b, out, nullptr);
}
